// RPN_34256659153102
// MI455X (gfx1250) — matrix operand order repaired and checked by reading, not compiled, not run on hardware
//
#include <hip/hip_runtime.h>
#include <hip/hip_bf16.h>
#include <math.h>

typedef __attribute__((ext_vector_type(16))) _Float16     v16h;
typedef __attribute__((ext_vector_type(8)))  float        v8f;
typedef __attribute__((ext_vector_type(4)))  unsigned int v4u;
typedef __attribute__((ext_vector_type(8)))  unsigned int v8u;

#define IOU_THR 0.3f
#define NBOX    9216          // 32*32*9
#define NSORT   16384         // next pow2
#define NW      288           // NBOX/32 bitmask words
#define KSTEPS  72            // 2304/32

union FragU { uint4 q[2]; v16h v; };

// ---------------------------------------------------------------------------
// TDM: 2D tile load Global->LDS via inline asm (portable across toolchains).
// Group1 (tensor shape) is constant; group0 carries lds/global addresses.
// ---------------------------------------------------------------------------
__device__ __forceinline__ void tdm_load_2d(unsigned lds_addr,
                                            unsigned long long gaddr,
                                            v8u g1)
{
  v4u g0;
  g0.x = 0x1u;                                   // count=1, user descriptor
  g0.y = lds_addr;                               // LDS byte address
  g0.z = (unsigned)gaddr;                        // global_addr[31:0]
  g0.w = (unsigned)(gaddr >> 32) | (2u << 30);   // global_addr[56:32] | type=2
  asm volatile("tensor_load_to_lds %0, %1" :: "s"(g0), "s"(g1) : "memory");
}

// ---------------------------------------------------------------------------
// Prep kernels: split f32 -> f16 hi + f16 lo (Markidis split), done ONCE so
// the WMMA hot loop has zero conversion VALU.
// ---------------------------------------------------------------------------
__global__ void prep_split_w(const float* __restrict__ W,
                             _Float16* __restrict__ WH,
                             _Float16* __restrict__ WL)
{
  int t = blockIdx.x * 256 + threadIdx.x;
  if (t >= 512 * 2304) return;
  float v = W[t];
  _Float16 hi = (_Float16)v;
  WH[t] = hi;
  WL[t] = (_Float16)(v - (float)hi);
}

// padded 34x34 feature maps with zero border: F[c][y+1][x+1] = feat[c][y][x]
__global__ void prep_split_feat(const float* __restrict__ feat,
                                _Float16* __restrict__ FH,
                                _Float16* __restrict__ FL)
{
  int t = blockIdx.x * 256 + threadIdx.x;
  if (t >= 256 * 34 * 34) return;
  int c = t / 1156, rem = t % 1156;
  int y = rem / 34, x = rem % 34;
  float v = 0.0f;
  if (y >= 1 && y <= 32 && x >= 1 && x <= 32)
    v = feat[c * 1024 + (y - 1) * 32 + (x - 1)];
  _Float16 hi = (_Float16)v;
  FH[t] = hi;
  FL[t] = (_Float16)(v - (float)hi);
}

// ---------------------------------------------------------------------------
// Kernel 1: 3x3 conv (pad=1) + bias + ReLU as im2col GEMM via WMMA f16 hi/lo.
// M=512 K=2304 N=1024. Block = 8 waves: 8 M-tiles x 1 N-tile.
// Software-pipelined: TDM DMA + B-stage for tile i+1 issued before the WMMAs
// of tile i; s_wait_tensorcnt only at the iteration boundary (latency hidden).
// ---------------------------------------------------------------------------
__global__ __launch_bounds__(256) void conv3x3_wmma(
    const _Float16* __restrict__ WH, const _Float16* __restrict__ WL,
    const _Float16* __restrict__ FH, const _Float16* __restrict__ FL,
    const float* __restrict__ bias, float* __restrict__ xout)
{
  const int n0   = blockIdx.x * 16;
  const int mg   = blockIdx.y;
  const int tid  = threadIdx.x;
  const int wave = tid >> 5;
  const int lane = tid & 31;
  const int g    = lane >> 4;
  const int ln   = lane & 15;

  __shared__ alignas(16) _Float16 Ahi[2][128][32];   // TDM-filled, dbl-buffered
  __shared__ alignas(16) _Float16 Alo[2][128][32];
  __shared__ alignas(16) _Float16 Bhi[2][16][32];    // [n][k]
  __shared__ alignas(16) _Float16 Blo[2][16][32];

  // constant TDM group1: data_size=2B, tensor 2304x512 stride 2304, tile 32x128
  v8u g1;
  g1[0] = 1u << 16;                        // data_size = 1 (2 bytes)
  g1[1] = (2304u & 0xFFFFu) << 16;         // tensor_dim0 low16
  g1[2] = (2304u >> 16) | (512u << 16);    // dim0 hi16 | tensor_dim1 lo16
  g1[3] = (512u >> 16) | (32u << 16);      // dim1 hi16 | tile_dim0 = 32
  g1[4] = 128u;                            // tile_dim1 = 128, tile_dim2 = 0
  g1[5] = 2304u;                           // tensor_dim0_stride low32
  g1[6] = 0u;                              // stride hi16 | dim1_stride lo16
  g1[7] = 0u;

  const unsigned long long whB = (unsigned long long)(size_t)WH +
                                 (unsigned long long)(mg * 128) * 2304ull * 2ull;
  const unsigned long long wlB = (unsigned long long)(size_t)WL +
                                 (unsigned long long)(mg * 128) * 2304ull * 2ull;
  const unsigned ldsAhi = (unsigned)(size_t)&Ahi[0][0][0];
  const unsigned ldsAlo = (unsigned)(size_t)&Alo[0][0][0];

  const int h  = n0 >> 5;                  // all 16 cols share image row
  const int w0 = n0 & 31;
  const int row = wave * 16 + ln;          // A-tile local row for frag reads

  // stage tile `it` into buffer `buf`: lanes copy im2col B, TDM DMAs A
  auto stage = [&](int it, int buf) {
    const int k0 = it * 32;
    #pragma unroll
    for (int q = 0; q < 2; ++q) {
      int e  = tid * 2 + q;
      int kk = e >> 4, nn = e & 15;
      int k  = k0 + kk;
      int ci = k / 9, rem = k % 9;
      int kh = rem / 3, kw = rem % 3;
      int ga = ci * 1156 + (h + kh) * 34 + (w0 + nn + kw);
      Bhi[buf][nn][kk] = FH[ga];
      Blo[buf][nn][kk] = FL[ga];
    }
    if (wave == 0) {
      unsigned off = (unsigned)buf * (128u * 32u * 2u);
      tdm_load_2d(ldsAhi + off, whB + (unsigned long long)k0 * 2ull, g1);
      tdm_load_2d(ldsAlo + off, wlB + (unsigned long long)k0 * 2ull, g1);
    }
  };

  v8f acc = {};

  // prologue: fill buffer 0
  stage(0, 0);
  if (wave == 0) __builtin_amdgcn_s_wait_tensorcnt(0);
  __syncthreads();

  for (int it = 0; it < KSTEPS; ++it) {
    const int cur = it & 1;
    if (it + 1 < KSTEPS) stage(it + 1, cur ^ 1);   // async: overlaps WMMAs

    // fragments: ISA layouts -> two contiguous 16B runs each (ds_load_b128)
    FragU ah, al, bh, bl;
    ah.q[0] = *(const uint4*)&Ahi[cur][row][g * 8];
    ah.q[1] = *(const uint4*)&Ahi[cur][row][16 + g * 8];
    al.q[0] = *(const uint4*)&Alo[cur][row][g * 8];
    al.q[1] = *(const uint4*)&Alo[cur][row][16 + g * 8];
    bh.q[0] = *(const uint4*)&Bhi[cur][ln][g * 8];
    bh.q[1] = *(const uint4*)&Bhi[cur][ln][16 + g * 8];
    bl.q[0] = *(const uint4*)&Blo[cur][ln][g * 8];
    bl.q[1] = *(const uint4*)&Blo[cur][ln][16 + g * 8];

    // split-precision accumulate: hi*hi + hi*lo + lo*hi (lo*lo ~2^-22, dropped)
    acc = __builtin_amdgcn_wmma_f32_16x16x32_f16(false, ah.v, false, bh.v,
                                                 (short)0, acc, false, false);
    acc = __builtin_amdgcn_wmma_f32_16x16x32_f16(false, ah.v, false, bl.v,
                                                 (short)0, acc, false, false);
    acc = __builtin_amdgcn_wmma_f32_16x16x32_f16(false, al.v, false, bh.v,
                                                 (short)0, acc, false, false);

    if (wave == 0) __builtin_amdgcn_s_wait_tensorcnt(0);  // next tile's DMA
    __syncthreads();   // also orders LDS reads (dscnt) vs next overwrite
  }

  // C layout: VGPR r, lanes 0-15 -> M=r, lanes 16-31 -> M=r+8
  #pragma unroll
  for (int r = 0; r < 8; ++r) {
    int mo = mg * 128 + wave * 16 + r + 8 * g;
    float v = acc[r] + bias[mo];
    xout[mo * 1024 + n0 + ln] = v > 0.0f ? v : 0.0f;
  }
}

// ---------------------------------------------------------------------------
// Kernel 2: 1x1 heads. c<18 -> sigmoid(logits) into probs (d_out region 2),
//           c in [18,54) -> raw bbox deltas into ws.
// ---------------------------------------------------------------------------
__global__ __launch_bounds__(256) void heads_1x1(
    const float* __restrict__ x, const float* __restrict__ cls_w,
    const float* __restrict__ cls_b, const float* __restrict__ bbox_w,
    const float* __restrict__ bbox_b, float* __restrict__ probs_out,
    float* __restrict__ bbox_out)
{
  int t = blockIdx.x * 256 + threadIdx.x;
  if (t >= 54 * 1024) return;
  int c = t >> 10, n = t & 1023;
  bool is_cls = c < 18;
  int cc = is_cls ? c : c - 18;
  const float* wrow = is_cls ? cls_w + cc * 512 : bbox_w + cc * 512;
  float accv = is_cls ? cls_b[cc] : bbox_b[cc];
  for (int i = 0; i < 512; ++i)
    accv = fmaf(wrow[i], x[i * 1024 + n], accv);
  if (is_cls) probs_out[cc * 1024 + n] = 1.0f / (1.0f + expf(-accv));
  else        bbox_out[cc * 1024 + n] = accv;
}

// ---------------------------------------------------------------------------
// Kernel 3: anchors (inline, round-half-even) + box decode + clip.
// Replicates the reference's channel-major reshape(-1,4) delta pairing.
// ---------------------------------------------------------------------------
__global__ void decode_boxes(const float* __restrict__ bbox,
                             const int* __restrict__ img,
                             float* __restrict__ prop)
{
  int r = blockIdx.x * 256 + threadIdx.x;
  if (r >= NBOX) return;
  const float scales[3] = {128.0f, 256.0f, 512.0f};
  const float ratios[3] = {0.5f, 1.0f, 2.0f};
  float img_sz = (float)img[0];
  float stride = (float)(img[0] / 32);
  int p = r / 9, a = r % 9;
  int gy = p >> 5, gx = p & 31;
  float hr = sqrtf(ratios[a / 3]);
  float wsz = (1.0f / hr) * scales[a % 3];
  float hsz = hr * scales[a % 3];
  float sx = gx * stride, sy = gy * stride;
  float ax1 = sx + rintf(-wsz * 0.5f), ay1 = sy + rintf(-hsz * 0.5f);
  float ax2 = sx + rintf( wsz * 0.5f), ay2 = sy + rintf( hsz * 0.5f);
  float aw = ax2 - ax1, ah = ay2 - ay1;
  float cx = ax1 + 0.5f * aw, cy = ay1 + 0.5f * ah;
  int f = 4 * r, c = f >> 10, pos = f & 1023;
  float dx = bbox[c * 1024 + pos],     dy = bbox[c * 1024 + pos + 1];
  float dw = bbox[c * 1024 + pos + 2], dh = bbox[c * 1024 + pos + 3];
  float pcx = dx * aw + cx, pcy = dy * ah + cy;
  float pw = expf(dw) * aw, ph = expf(dh) * ah;
  float x1 = fminf(fmaxf(pcx - 0.5f * pw, 0.0f), img_sz);
  float y1 = fminf(fmaxf(pcy - 0.5f * ph, 0.0f), img_sz);
  float x2 = fminf(fmaxf(pcx + 0.5f * pw, 0.0f), img_sz);
  float y2 = fminf(fmaxf(pcy + 0.5f * ph, 0.0f), img_sz);
  prop[r * 4] = x1; prop[r * 4 + 1] = y1; prop[r * 4 + 2] = x2; prop[r * 4 + 3] = y2;
}

// ---------------------------------------------------------------------------
// NMS: stable argsort(-score) via bitonic sort on packed u64 keys
// ---------------------------------------------------------------------------
__global__ void nms_keys_init(const float* __restrict__ probs,
                              unsigned long long* __restrict__ keys)
{
  int i = blockIdx.x * 256 + threadIdx.x;
  if (i >= NSORT) return;
  unsigned long long key = ~0ull;                 // pad sorts last
  if (i < NBOX) {
    int ch = 1 + 2 * (i >> 10);                   // probs[:,1::2] flatten order
    float s = probs[ch * 1024 + (i & 1023)];
    unsigned u = __float_as_uint(s);
    u = (u & 0x80000000u) ? ~u : (u | 0x80000000u); // ascending-sortable
    u = ~u;                                         // descending score
    key = ((unsigned long long)u << 32) | (unsigned)i; // tie: smaller idx first
  }
  keys[i] = key;
}

__global__ void bitonic_step(unsigned long long* __restrict__ keys, int k, int j)
{
  int i = blockIdx.x * 256 + threadIdx.x;
  int ixj = i ^ j;
  if (ixj > i) {
    unsigned long long a = keys[i], b = keys[ixj];
    bool up = ((i & k) == 0);
    if ((a > b) == up) { keys[i] = b; keys[ixj] = a; }
  }
}

__global__ void gather_sorted(const unsigned long long* __restrict__ keys,
                              const float* __restrict__ prop,
                              float* __restrict__ bs, float* __restrict__ area)
{
  int i = blockIdx.x * 256 + threadIdx.x;
  if (i >= NBOX) return;
  int o = (int)(keys[i] & 0xffffffffu);
  float4 p = *(const float4*)(prop + o * 4);
  *(float4*)(bs + i * 4) = p;
  area[i] = (p.z - p.x) * (p.w - p.y);
}

// parallel suppression bitmask: mat[i][w] bit b = (iou(i, 32w+b)>thr && j>i)
__global__ void supp_matrix(const float* __restrict__ bs,
                            const float* __restrict__ area,
                            unsigned* __restrict__ mat)
{
  int t = blockIdx.x * 256 + threadIdx.x;
  if (t >= NBOX * NW) return;
  int i = t / NW, wb = t % NW;
  float ix1 = bs[i*4], iy1 = bs[i*4+1], ix2 = bs[i*4+2], iy2 = bs[i*4+3];
  float ia = area[i];
  unsigned bits = 0;
  #pragma unroll 4
  for (int b = 0; b < 32; ++b) {
    int j = wb * 32 + b;
    float x1 = fmaxf(ix1, bs[j*4]);
    float y1 = fmaxf(iy1, bs[j*4+1]);
    float x2 = fminf(ix2, bs[j*4+2]);
    float y2 = fminf(iy2, bs[j*4+3]);
    float inter = fmaxf(x2 - x1, 0.0f) * fmaxf(y2 - y1, 0.0f);
    float iou = inter / (ia + area[j] - inter + 1e-9f);
    if (iou > IOU_THR && j > i) bits |= (1u << b);
  }
  mat[(size_t)i * NW + wb] = bits;
}

// single-block serial scan over sorted order (matches fori_loop semantics)
__global__ __launch_bounds__(512) void nms_scan(const unsigned* __restrict__ mat,
                                                unsigned* __restrict__ keepw)
{
  __shared__ unsigned keep[NW];
  int t = threadIdx.x;
  for (int w = t; w < NW; w += 512) keep[w] = 0xffffffffu;
  __syncthreads();
  for (int i = 0; i < NBOX; ++i) {
    unsigned wv = keep[i >> 5];          // uniform read
    __syncthreads();
    if ((wv >> (i & 31)) & 1) {
      const unsigned* row = mat + (size_t)i * NW;
      for (int w = t; w < NW; w += 512) keep[w] &= ~row[w];
    }
    __syncthreads();
  }
  for (int w = t; w < NW; w += 512) keepw[w] = keep[w];
}

// scatter to original order; write proposals*keep and keep to d_out
__global__ void nms_finalize(const unsigned long long* __restrict__ keys,
                             const unsigned* __restrict__ keepw,
                             const float* __restrict__ prop,
                             float* __restrict__ out)
{
  int i = blockIdx.x * 256 + threadIdx.x;
  if (i >= NBOX) return;
  int o = (int)(keys[i] & 0xffffffffu);
  float k = ((keepw[i >> 5] >> (i & 31)) & 1) ? 1.0f : 0.0f;
  float4 p = *(const float4*)(prop + o * 4);
  out[o * 4]     = p.x * k;
  out[o * 4 + 1] = p.y * k;
  out[o * 4 + 2] = p.z * k;
  out[o * 4 + 3] = p.w * k;
  out[NBOX * 4 + 18 * 1024 + o] = k;      // keep mask region
}

// ---------------------------------------------------------------------------
extern "C" void kernel_launch(void* const* d_in, const int* in_sizes, int n_in,
                              void* d_out, int out_size, void* d_ws, size_t ws_size,
                              hipStream_t stream) {
  const float* feat   = (const float*)d_in[0];
  const int*   imgsz  = (const int*)  d_in[1];
  const float* conv_w = (const float*)d_in[2];
  const float* conv_b = (const float*)d_in[3];
  const float* cls_w  = (const float*)d_in[4];
  const float* cls_b  = (const float*)d_in[5];
  const float* bbox_w = (const float*)d_in[6];
  const float* bbox_b = (const float*)d_in[7];
  float* out = (float*)d_out;

  char* ws = (char*)d_ws;
  float*              X    = (float*)             (ws + 0);          // 2 MB
  float*              BB   = (float*)             (ws + 2097152);    // 144 KB
  float*              PROP = (float*)             (ws + 2244608);    // 144 KB
  unsigned long long* KEYS = (unsigned long long*)(ws + 2392064);    // 128 KB
  float*              BS   = (float*)             (ws + 2523136);    // 144 KB
  float*              AREA = (float*)             (ws + 2670592);    // 36 KB
  unsigned*           MAT  = (unsigned*)          (ws + 2707456);    // 10.6 MB
  unsigned*           KEEP = (unsigned*)          (ws + 13324288);   // 1.2 KB
  _Float16*           WH   = (_Float16*)          (ws + 13325568);   // 2.25 MB
  _Float16*           WL   = (_Float16*)          (ws + 15684864);   // 2.25 MB
  _Float16*           FH   = (_Float16*)          (ws + 18044160);   // 578 KB
  _Float16*           FL   = (_Float16*)          (ws + 18636032);   // 578 KB

  float* probs_out = out + NBOX * 4;   // d_out layout: boxes | probs | keep

  prep_split_w<<<4608, 256, 0, stream>>>(conv_w, WH, WL);
  prep_split_feat<<<1156, 256, 0, stream>>>(feat, FH, FL);
  conv3x3_wmma<<<dim3(64, 4), 256, 0, stream>>>(WH, WL, FH, FL, conv_b, X);
  heads_1x1<<<216, 256, 0, stream>>>(X, cls_w, cls_b, bbox_w, bbox_b,
                                     probs_out, BB);
  decode_boxes<<<36, 256, 0, stream>>>(BB, imgsz, PROP);
  nms_keys_init<<<64, 256, 0, stream>>>(probs_out, KEYS);
  for (int k = 2; k <= NSORT; k <<= 1)
    for (int j = k >> 1; j > 0; j >>= 1)
      bitonic_step<<<64, 256, 0, stream>>>(KEYS, k, j);
  gather_sorted<<<36, 256, 0, stream>>>(KEYS, PROP, BS, AREA);
  supp_matrix<<<(NBOX * NW + 255) / 256, 256, 0, stream>>>(BS, AREA, MAT);
  nms_scan<<<1, 512, 0, stream>>>(MAT, KEEP);
  nms_finalize<<<36, 256, 0, stream>>>(KEYS, KEEP, PROP, out);
}
